// LandmarkMemory_32641751449695
// MI455X (gfx1250) — matrix operand order repaired and checked by reading, not compiled, not run on hardware
//
#include <hip/hip_runtime.h>
#include <hip/hip_bf16.h>

typedef __attribute__((ext_vector_type(16))) _Float16 v16h;
typedef __attribute__((ext_vector_type(8)))  _Float16 v8h;
typedef __attribute__((ext_vector_type(8)))  float    v8f;

#define BB 4
#define LL 8192
#define DD 1024
#define MM 2000
#define MP 2016          // M padded to multiple of 32
#define POOLR 15
#define STHRESH 0.15f
#define RSCALE 5.0f
#define MAXIDX 2048

// ---------- fragment load helpers (wave32 WMMA layouts, ISA 7.12.2) ----------

// A-matrix 16x32 f16 fragment from a row-major row pointer:
// lane<16: K = k0 + {0..15}; lane>=16: K = k0 + {16..31}
__device__ __forceinline__ v16h ld_a_contig(const _Float16* rowptr, int k0, bool hi) {
  const _Float16* p = rowptr + k0 + (hi ? 16 : 0);
  v8h a0 = *(const v8h*)(p);
  v8h a1 = *(const v8h*)(p + 8);
  v16h a;
#pragma unroll
  for (int j = 0; j < 8; ++j) { a[j] = a0[j]; a[8 + j] = a1[j]; }
  return a;
}

// B-matrix 32x16 f16 fragment where this lane's 16 K-values are contiguous
// (data stored N-major: element (k,n) at base[n*ld + k])
__device__ __forceinline__ v16h ld_b_contig(const _Float16* p) {
  v8h b0 = *(const v8h*)(p);
  v8h b1 = *(const v8h*)(p + 8);
  v16h b;
#pragma unroll
  for (int j = 0; j < 8; ++j) { b[j] = b0[j]; b[8 + j] = b1[j]; }
  return b;
}

// ------------------------------- prep kernels -------------------------------

__global__ __launch_bounds__(256)
void prep_tokens(const float* __restrict__ x, const float* __restrict__ Wunc,
                 const float* __restrict__ bunc, const float* __restrict__ Wsal,
                 const float* __restrict__ bsal, _Float16* __restrict__ qf16,
                 float* __restrict__ need, float* __restrict__ sal)
{
  int token = blockIdx.x * 8 + (threadIdx.x >> 5);
  int lane  = threadIdx.x & 31;
  const float* xr = x + (size_t)token * DD;
  float ss = 0.f, du = 0.f, dsal = 0.f;
  for (int i = lane; i < DD; i += 32) {
    float v = xr[i];
    ss += v * v; du += v * Wunc[i]; dsal += v * Wsal[i];
  }
#pragma unroll
  for (int o = 16; o; o >>= 1) {
    ss += __shfl_xor(ss, o, 32);
    du += __shfl_xor(du, o, 32);
    dsal += __shfl_xor(dsal, o, 32);
  }
  float inv = 1.0f / fmaxf(sqrtf(ss), 1e-12f);
  if (lane == 0) {
    need[token] = 1.f / (1.f + __expf(-(du + bunc[0])));
    sal[token]  = 1.f / (1.f + __expf(-(dsal + bsal[0])));
  }
  for (int i = lane; i < DD; i += 32)
    qf16[(size_t)token * DD + i] = (_Float16)(xr[i] * inv);
}

// kf16: [b][m][d] row-major (B-fragment contiguous for Q*K^T)
// vf16T: [b][d][m] (M-stride MP) -> B-fragment contiguous for P*V
__global__ __launch_bounds__(256)
void prep_landmarks(const float* __restrict__ lk, const float* __restrict__ lv,
                    _Float16* __restrict__ kf16, _Float16* __restrict__ vf16T)
{
  int idx  = blockIdx.x * 8 + (threadIdx.x >> 5);
  int lane = threadIdx.x & 31;
  int b = idx / MP, m = idx % MP;
  _Float16* ko = kf16 + ((size_t)b * MP + m) * DD;
  _Float16* vo = vf16T + (size_t)b * DD * MP + m;
  if (m < MM) {
    const float* kr = lk + ((size_t)b * MM + m) * DD;
    const float* vr = lv + ((size_t)b * MM + m) * DD;
    float ss = 0.f;
    for (int i = lane; i < DD; i += 32) { float v = kr[i]; ss += v * v; }
#pragma unroll
    for (int o = 16; o; o >>= 1) ss += __shfl_xor(ss, o, 32);
    float inv = 1.0f / fmaxf(sqrtf(ss), 1e-12f);
    for (int i = lane; i < DD; i += 32) {
      ko[i] = (_Float16)(kr[i] * inv);
      vo[(size_t)i * MP] = (_Float16)vr[i];
    }
  } else {
    for (int i = lane; i < DD; i += 32) {
      ko[i] = (_Float16)0.f;
      vo[(size_t)i * MP] = (_Float16)0.f;
    }
  }
}

// W16T[n][k] = W_mem[k][n]  (B-fragment contiguous for r @ W_mem)
__global__ __launch_bounds__(256)
void prep_wmem(const float* __restrict__ W, _Float16* __restrict__ W16T)
{
  for (int i = blockIdx.x * 256 + threadIdx.x; i < DD * DD; i += gridDim.x * 256) {
    int n = i / DD, k = i % DD;
    W16T[i] = (_Float16)W[(size_t)k * DD + n];
  }
}

// --------------------------- fused attention kernel --------------------------
// one block = 16 tokens, 8 waves
// LDS: S[16][2016] f32 (129KB) + P16[16][2016] f16 (63KB) + qt[16][1024] f16 (32KB)

__global__ __launch_bounds__(256)
void attn_fused(const _Float16* __restrict__ qf16, const _Float16* __restrict__ kf16,
                const _Float16* __restrict__ vf16T, const _Float16* __restrict__ wm16T,
                const float* __restrict__ need_g, const float* __restrict__ x,
                const float* __restrict__ b_mem, float* __restrict__ xout)
{
  extern __shared__ char smem[];
  float*    S   = (float*)smem;                                  // 16*2016 f32
  _Float16* P16 = (_Float16*)(smem + 16 * MP * sizeof(float));   // 16*2016 f16
  _Float16* qt  = (_Float16*)(smem + 16 * MP * 6);               // 16*1024 f16 (q, then r)

  const int  tid  = threadIdx.x;
  const int  wave = tid >> 5;
  const int  lane = tid & 31;
  const int  row  = lane & 15;
  const bool hi   = lane >= 16;

  const int bb = blockIdx.x >> 9;        // /(L/16)
  const int t0 = (blockIdx.x & 511) << 4;

  // stage q tile
  const _Float16* qg = qf16 + ((size_t)bb * LL + t0) * DD;
  for (int i = tid; i < 16 * DD / 8; i += 256)
    ((v8h*)qt)[i] = ((const v8h*)qg)[i];
  __syncthreads();

  // -------- Phase 1: S = RSCALE * Qn Kn^T --------
  const _Float16* kb = kf16 + (size_t)bb * MP * DD;
  for (int tile = wave; tile < MP / 16; tile += 8) {
    int n0 = tile * 16;
    const _Float16* krow = kb + (size_t)(n0 + row) * DD + (hi ? 16 : 0);
    v8f c = {};
    for (int ks = 0; ks < DD / 32; ++ks) {
      int k0 = ks * 32;
      v16h a = ld_a_contig(qt + row * DD, k0, hi);
      v16h b = ld_b_contig(krow + k0);
      c = __builtin_amdgcn_wmma_f32_16x16x32_f16(false, a, false, b, (short)0, c, false, false);
    }
    int col = n0 + row;
#pragma unroll
    for (int r = 0; r < 8; ++r) {
      int mrow = r + (hi ? 8 : 0);
      S[mrow * MP + col] = (col < MM) ? RSCALE * c[r] : -1e30f;
    }
  }
  __syncthreads();

  // -------- Phase 2: softmax along M; write P16 (contiguous f16) ----
  {
    int tok = tid >> 4, sub = tid & 15;
    float* Sr = S + tok * MP;
    _Float16* Pr = P16 + tok * MP;
    float mx = -1e30f;
    for (int i = sub; i < MP; i += 16) mx = fmaxf(mx, Sr[i]);
#pragma unroll
    for (int o = 8; o; o >>= 1) mx = fmaxf(mx, __shfl_xor(mx, o, 16));
    float sum = 0.f;
    for (int i = sub; i < MP; i += 16) sum += __expf(Sr[i] - mx);
#pragma unroll
    for (int o = 8; o; o >>= 1) sum += __shfl_xor(sum, o, 16);
    float inv = 1.0f / sum;
    for (int i = sub; i < MP; i += 16)
      Pr[i] = (_Float16)(__expf(Sr[i] - mx) * inv);
  }
  __syncthreads();

  // -------- Phase 3: r = need * (P @ V) -> qt (f16) --------
  const _Float16* vbT = vf16T + (size_t)bb * DD * MP;
  float nd[8];
#pragma unroll
  for (int r = 0; r < 8; ++r)
    nd[r] = need_g[(size_t)bb * LL + t0 + r + (hi ? 8 : 0)];

  for (int nt = 0; nt < 8; ++nt) {
    int n0 = (wave << 7) + (nt << 4);
    const _Float16* vrow = vbT + (size_t)(n0 + row) * MP + (hi ? 16 : 0);
    v8f c = {};
    for (int ks = 0; ks < MP / 32; ++ks) {
      int k0 = ks * 32;
      v16h a = ld_a_contig(P16 + row * MP, k0, hi);
      v16h b = ld_b_contig(vrow + k0);
      c = __builtin_amdgcn_wmma_f32_16x16x32_f16(false, a, false, b, (short)0, c, false, false);
    }
#pragma unroll
    for (int r = 0; r < 8; ++r) {
      int mrow = r + (hi ? 8 : 0);
      qt[mrow * DD + n0 + row] = (_Float16)(nd[r] * c[r]);
    }
  }
  __syncthreads();

  // -------- Phase 4: x_out = x + r @ W_mem + b_mem --------
  const float* xg = x    + ((size_t)bb * LL + t0) * DD;
  float*       xo = xout + ((size_t)bb * LL + t0) * DD;
  for (int nt = 0; nt < 8; ++nt) {
    int n0 = (wave << 7) + (nt << 4);
    const _Float16* wrow = wm16T + (size_t)(n0 + row) * DD + (hi ? 16 : 0);
    v8f c = {};
    for (int ks = 0; ks < DD / 32; ++ks) {
      int k0 = ks * 32;
      v16h a = ld_a_contig(qt + row * DD, k0, hi);
      v16h b = ld_b_contig(wrow + k0);
      c = __builtin_amdgcn_wmma_f32_16x16x32_f16(false, a, false, b, (short)0, c, false, false);
    }
#pragma unroll
    for (int r = 0; r < 8; ++r) {
      int mrow = r + (hi ? 8 : 0);
      int col  = n0 + row;
      xo[(size_t)mrow * DD + col] = xg[(size_t)mrow * DD + col] + c[r] + b_mem[col];
    }
  }
}

// ----------------------- landmark detection & update -------------------------

__global__ __launch_bounds__(1024)
void find_landmarks(const float* __restrict__ sal, int* __restrict__ idx,
                    int* __restrict__ cnt, float* __restrict__ penalty)
{
  __shared__ float s_sal[LL];
  __shared__ unsigned char s_flag[LL];
  int tid = threadIdx.x;
  for (int i = tid; i < LL; i += 1024) s_sal[i] = sal[i];   // batch 0
  __syncthreads();
  for (int i = tid; i < LL; i += 1024) {
    float v = s_sal[i], mx = -1e30f;
#pragma unroll 4
    for (int w = -POOLR; w <= POOLR; ++w) {
      int p = i + w;
      float q = (p >= 0 && p < LL) ? s_sal[p] : -1e30f;
      mx = fmaxf(mx, q);
    }
    s_flag[i] = (v == mx && v > STHRESH) ? 1 : 0;
  }
  __syncthreads();
  if (tid == 0) {
    int c = 0;
    for (int i = 0; i < LL; ++i)
      if (s_flag[i]) { if (c < MAXIDX) idx[c] = i; ++c; }
    cnt[0] = c < MAXIDX ? c : MAXIDX;
    penalty[0] = 0.f;
  }
}

__global__ __launch_bounds__(256)
void update_landmarks(const float* __restrict__ x, const float* __restrict__ xout,
                      const float* __restrict__ lk, const float* __restrict__ lv,
                      const float* __restrict__ Wwr, const float* __restrict__ bwr,
                      const int* __restrict__ idx, const int* __restrict__ cnt,
                      float* __restrict__ outk, float* __restrict__ outv,
                      float* __restrict__ penalty)
{
  __shared__ float red[256];
  int blk = blockIdx.x, b = blk / MM, m = blk % MM;
  int tid = threadIdx.x;
  int count = cnt[0];
  int keep = MM - count;
  float* dk = outk + ((size_t)b * MM + m) * DD;
  float* dv = outv + ((size_t)b * MM + m) * DD;
  if (m < keep) {
    const float* sk = lk + ((size_t)b * MM + m + count) * DD;
    const float* sv = lv + ((size_t)b * MM + m + count) * DD;
    for (int i = tid; i < DD; i += 256) { dk[i] = sk[i]; dv[i] = sv[i]; }
  } else {
    int j = m - keep;
    int t = idx[j];
    const float* xr = x + ((size_t)b * LL + t) * DD;
    float d = 0.f;
    for (int i = tid; i < DD; i += 256) d += xr[i] * Wwr[i];
    red[tid] = d; __syncthreads();
    for (int s = 128; s; s >>= 1) { if (tid < s) red[tid] += red[tid + s]; __syncthreads(); }
    float gate = 1.f / (1.f + __expf(-(red[0] + bwr[0])));
    const float* xo = xout + ((size_t)b * LL + t) * DD;
    for (int i = tid; i < DD; i += 256) { dk[i] = xr[i]; dv[i] = xo[i] * gate; }
    if (tid == 0)
      atomicAdd(penalty, (1.f - gate) * (1.f - gate) / ((float)BB * (float)count));
  }
}

// --------------------------------- launcher ----------------------------------

extern "C" void kernel_launch(void* const* d_in, const int* in_sizes, int n_in,
                              void* d_out, int out_size, void* d_ws, size_t ws_size,
                              hipStream_t stream)
{
  const float* x    = (const float*)d_in[0];
  const float* lk   = (const float*)d_in[1];
  const float* lv   = (const float*)d_in[2];
  const float* Wsal = (const float*)d_in[3];
  const float* bsal = (const float*)d_in[4];
  const float* Wunc = (const float*)d_in[5];
  const float* bunc = (const float*)d_in[6];
  const float* Wwr  = (const float*)d_in[7];
  const float* bwr  = (const float*)d_in[8];
  const float* Wmem = (const float*)d_in[9];
  const float* bmem = (const float*)d_in[10];

  float* out  = (float*)d_out;
  float* xout = out;                                     // B*L*D
  float* outk = xout + (size_t)BB * LL * DD;             // B*M*D
  float* outv = outk + (size_t)BB * MM * DD;             // B*M*D
  float* pen  = outv + (size_t)BB * MM * DD;             // 1

  char* ws = (char*)d_ws;
  size_t o = 0;
  _Float16* qf16  = (_Float16*)(ws + o); o += (size_t)BB * LL * DD * 2;
  _Float16* kf16  = (_Float16*)(ws + o); o += (size_t)BB * MP * DD * 2;
  _Float16* vf16T = (_Float16*)(ws + o); o += (size_t)BB * MP * DD * 2;
  _Float16* wm16T = (_Float16*)(ws + o); o += (size_t)DD * DD * 2;
  float* need     = (float*)(ws + o);    o += (size_t)BB * LL * 4;
  float* sal      = (float*)(ws + o);    o += (size_t)BB * LL * 4;
  int* idx        = (int*)(ws + o);      o += MAXIDX * 4;
  int* cnt        = (int*)(ws + o);      o += 64;

  prep_tokens<<<BB * LL / 8, 256, 0, stream>>>(x, Wunc, bunc, Wsal, bsal, qf16, need, sal);
  prep_landmarks<<<BB * MP / 8, 256, 0, stream>>>(lk, lv, kf16, vf16T);
  prep_wmem<<<512, 256, 0, stream>>>(Wmem, wm16T);

  size_t smem = (size_t)16 * MP * 6 + (size_t)16 * DD * sizeof(_Float16);
  attn_fused<<<BB * (LL / 16), 256, smem, stream>>>(qf16, kf16, vf16T, wm16T, need, x, bmem, xout);

  find_landmarks<<<1, 1024, 0, stream>>>(sal, idx, cnt, pen);
  update_landmarks<<<BB * MM, 256, 0, stream>>>(x, xout, lk, lv, Wwr, bwr, idx, cnt,
                                                outk, outv, pen);
}
